// HalfConv_876173328516
// MI455X (gfx1250) — matrix operand order repaired and checked by reading, not compiled, not run on hardware
//
#include <hip/hip_runtime.h>
#include <stdint.h>

typedef __attribute__((ext_vector_type(16))) __bf16 v16bf;
typedef __attribute__((ext_vector_type(8)))  float  v8f;

#define U_N 50000
#define E_N 800000

union FragBF {
  v16bf    v;
  __bf16   h[16];
  uint32_t u[8];
  uint4    q[2];
};

__device__ __forceinline__ uint32_t pack2(float a, float b) {
  union { __bf16 h[2]; uint32_t w; } c;
  c.h[0] = (__bf16)a;
  c.h[1] = (__bf16)b;
  return c.w;
}

// Load 8 consecutive f32 (32B aligned) -> 8 bf16 fragment elems (no branch).
__device__ __forceinline__ void pack_run(const float* __restrict__ p, __bf16* dst) {
  float4 a = ((const float4*)p)[0];
  float4 b = ((const float4*)p)[1];
  dst[0] = (__bf16)a.x; dst[1] = (__bf16)a.y;
  dst[2] = (__bf16)a.z; dst[3] = (__bf16)a.w;
  dst[4] = (__bf16)b.x; dst[5] = (__bf16)b.y;
  dst[6] = (__bf16)b.z; dst[7] = (__bf16)b.w;
}

__device__ __forceinline__ void pack_zero(__bf16* dst) {
  __bf16 z = (__bf16)0.0f;
  for (int i = 0; i < 8; ++i) dst[i] = z;
}

// B fragment from pre-swizzled weights: lane-contiguous 32B blocks.
__device__ __forceinline__ FragBF load_bfrag(const uint32_t* __restrict__ swz,
                                             int t, int s, int NK, int lane) {
  FragBF b;
  const uint4* p = (const uint4*)(swz + (((size_t)(t * NK + s) * 32 + lane) << 3));
  b.q[0] = p[0];
  b.q[1] = p[1];
  return b;
}

// ---------------------------------------------------------------------------
// Weight swizzle: out[((t*NK+s)*32+lane)*8 + vv] packs W[kb][col],W[kb+1][col]
// with col = t*16 + lane%16, kb = s*32 + (lane/16)*8 + 2*(vv%4) + 16*(vv/4). Rows >= Kreal
// are zero-padded (g_w1: K 144 -> 160).
// ---------------------------------------------------------------------------
__global__ void convert_weights(const float* __restrict__ gw1,
                                const float* __restrict__ gw2,
                                const float* __restrict__ fw1,
                                const float* __restrict__ fw2,
                                uint32_t* __restrict__ ws) {
  int p = blockIdx.x * blockDim.x + threadIdx.x;
  const float* W; uint32_t* out; int NK, Kreal, N, local;
  if (p < 10240)      { W = gw1; out = ws;         NK = 5; Kreal = 144; N = 128; local = p;         }
  else if (p < 14336) { W = gw2; out = ws + 10240; NK = 4; Kreal = 128; N = 64;  local = p - 10240; }
  else if (p < 22528) { W = fw1; out = ws + 14336; NK = 4; Kreal = 128; N = 128; local = p - 14336; }
  else if (p < 26624) { W = fw2; out = ws + 22528; NK = 4; Kreal = 128; N = 64;  local = p - 22528; }
  else return;
  int vv   = local & 7;
  int lane = (local >> 3) & 31;
  int ts   = local >> 8;
  int s    = ts % NK;
  int t    = ts / NK;
  int col  = t * 16 + (lane & 15);
  int kb   = s * 32 + (lane >> 4) * 8 + (vv & 3) * 2 + (vv >> 2) * 16;
  float f0 = (kb     < Kreal) ? W[(size_t)kb * N + col]       : 0.0f;
  float f1 = (kb + 1 < Kreal) ? W[(size_t)(kb + 1) * N + col] : 0.0f;
  out[local] = pack2(f0, f1);
}

__global__ void zero_f32(float* __restrict__ p, int n) {
  int i = blockIdx.x * blockDim.x + threadIdx.x;
  if (i < n) p[i] = 0.0f;
}

// ---------------------------------------------------------------------------
// Edge MLP + scatter-add. One wave per 16-edge tile, 4 waves per block.
// ---------------------------------------------------------------------------
__global__ __launch_bounds__(128) void edge_mlp(
    const float* __restrict__ u, const float* __restrict__ v,
    const float* __restrict__ ev, const int* __restrict__ idxV,
    const int* __restrict__ idxU,
    const uint32_t* __restrict__ gw1s, const uint32_t* __restrict__ gw2s,
    const float* __restrict__ gb1, const float* __restrict__ gb2,
    float* __restrict__ agg) {
  __shared__ __align__(16) __bf16 Hl[4][16 * 128];  // per-wave 16x128 bf16

  const int lane   = threadIdx.x & 31;
  const int wave   = threadIdx.x >> 5;
  const int tile   = blockIdx.x * 4 + wave;
  const int col16  = lane & 15;
  const int hiHalf = lane >> 4;
  const int edge   = tile * 16 + col16;   // this lane's edge row (M = col16)
  const int iu     = idxU[edge];
  const int iv     = idxV[edge];

  // Per-lane row base pointers (region statically known per K-step).
  const float* rowu = u  + (size_t)iu   * 64 + hiHalf * 8;
  const float* rowv = v  + (size_t)iv   * 64 + hiHalf * 8;
  const float* rowe = ev + (size_t)edge * 16 + hiHalf * 8;

  // ---- layer 1: X(16x160) @ gW1(160x128) -> 16x128 -------------------------
  v8f acc[8];
  for (int t = 0; t < 8; ++t)
    for (int r = 0; r < 8; ++r) acc[t][r] = 0.0f;

  auto step1 = [&](int s, const float* p0, const float* p1) {
    FragBF a;
    pack_run(p0, &a.h[0]);
    if (p1) pack_run(p1, &a.h[8]); else pack_zero(&a.h[8]);
    for (int t = 0; t < 8; ++t) {
      FragBF b = load_bfrag(gw1s, t, s, 5, lane);
      acc[t] = __builtin_amdgcn_wmma_f32_16x16x32_bf16(
          false, a.v, false, b.v, (short)0, acc[t], false, false);
    }
  };
  step1(0, rowu,      rowu + 16);   // K   0.. 31: u features
  step1(1, rowu + 32, rowu + 48);   // K  32.. 63: u features
  step1(2, rowv,      rowv + 16);   // K  64.. 95: v features
  step1(3, rowv + 32, rowv + 48);   // K  96..127: v features
  step1(4, rowe,      nullptr);     // K 128..143: e_vals, 144..159: zero pad

  // bias + ReLU, transpose to per-wave LDS as bf16 rows
  __bf16* h = &Hl[wave][0];
  for (int t = 0; t < 8; ++t) {
    float bias = gb1[t * 16 + col16];
    for (int r = 0; r < 8; ++r) {
      float val = fmaxf(acc[t][r] + bias, 0.0f);
      int row = r + hiHalf * 8;
      h[row * 128 + t * 16 + col16] = (__bf16)val;
    }
  }

  // ---- layer 2: H(16x128) @ gW2(128x64) -> 16x64 ----------------------------
  v8f acc2[4];
  for (int t = 0; t < 4; ++t)
    for (int r = 0; r < 8; ++r) acc2[t][r] = 0.0f;

  const __bf16* hrow = &Hl[wave][(size_t)col16 * 128];
  for (int s = 0; s < 4; ++s) {
    FragBF a;
    int kb0 = s * 32 + hiHalf * 8;
    a.q[0] = *(const uint4*)(hrow + kb0);
    a.q[1] = *(const uint4*)(hrow + kb0 + 16);
    for (int t = 0; t < 4; ++t) {
      FragBF b = load_bfrag(gw2s, t, s, 4, lane);
      acc2[t] = __builtin_amdgcn_wmma_f32_16x16x32_bf16(
          false, a.v, false, b.v, (short)0, acc2[t], false, false);
    }
  }

  // bias + ReLU + scatter-add (segment_sum over e_idx_u)
  for (int r = 0; r < 8; ++r) {
    int srcRow = r + hiHalf * 8;
    int dst = __shfl(iu, srcRow, 32);   // lane m holds iu for edge row m
    float* ap = agg + (size_t)dst * 64;
    for (int t = 0; t < 4; ++t) {
      float val = fmaxf(acc2[t][r] + gb2[t * 16 + col16], 0.0f);
      __hip_atomic_fetch_add(ap + t * 16 + col16, val,
                             __ATOMIC_RELAXED, __HIP_MEMORY_SCOPE_AGENT);
    }
  }
}

// ---------------------------------------------------------------------------
// Node MLP: out = ReLU(ReLU([u|agg] @ fW1 + b1) @ fW2 + b2). agg/out alias.
// ---------------------------------------------------------------------------
__global__ __launch_bounds__(128) void node_mlp(
    const float* __restrict__ u, const float* __restrict__ agg_in,
    const uint32_t* __restrict__ fw1s, const uint32_t* __restrict__ fw2s,
    const float* __restrict__ fb1, const float* __restrict__ fb2,
    float* __restrict__ out) {
  __shared__ __align__(16) __bf16 Hl[4][16 * 128];

  const int lane = threadIdx.x & 31;
  const int wave = threadIdx.x >> 5;
  const int tile = blockIdx.x * 4 + wave;
  if (tile >= U_N / 16) return;
  const int col16  = lane & 15;
  const int hiHalf = lane >> 4;
  const int row    = tile * 16 + col16;

  const float* rowu = u      + (size_t)row * 64 + hiHalf * 8;
  const float* rowa = agg_in + (size_t)row * 64 + hiHalf * 8;

  v8f acc[8];
  for (int t = 0; t < 8; ++t)
    for (int r = 0; r < 8; ++r) acc[t][r] = 0.0f;

  auto step1 = [&](int s, const float* p0) {
    FragBF a;
    pack_run(p0,      &a.h[0]);
    pack_run(p0 + 16, &a.h[8]);
    for (int t = 0; t < 8; ++t) {
      FragBF b = load_bfrag(fw1s, t, s, 4, lane);
      acc[t] = __builtin_amdgcn_wmma_f32_16x16x32_bf16(
          false, a.v, false, b.v, (short)0, acc[t], false, false);
    }
  };
  step1(0, rowu);        // K  0..31 : u
  step1(1, rowu + 32);   // K 32..63 : u
  step1(2, rowa);        // K 64..95 : agg
  step1(3, rowa + 32);   // K 96..127: agg

  __bf16* h = &Hl[wave][0];
  for (int t = 0; t < 8; ++t) {
    float bias = fb1[t * 16 + col16];
    for (int r = 0; r < 8; ++r) {
      float val = fmaxf(acc[t][r] + bias, 0.0f);
      int rr = r + hiHalf * 8;
      h[rr * 128 + t * 16 + col16] = (__bf16)val;
    }
  }

  v8f acc2[4];
  for (int t = 0; t < 4; ++t)
    for (int r = 0; r < 8; ++r) acc2[t][r] = 0.0f;

  const __bf16* hrow = &Hl[wave][(size_t)col16 * 128];
  for (int s = 0; s < 4; ++s) {
    FragBF a;
    int kb0 = s * 32 + hiHalf * 8;
    a.q[0] = *(const uint4*)(hrow + kb0);
    a.q[1] = *(const uint4*)(hrow + kb0 + 16);
    for (int t = 0; t < 4; ++t) {
      FragBF b = load_bfrag(fw2s, t, s, 4, lane);
      acc2[t] = __builtin_amdgcn_wmma_f32_16x16x32_bf16(
          false, a.v, false, b.v, (short)0, acc2[t], false, false);
    }
  }

  for (int r = 0; r < 8; ++r) {
    int node = tile * 16 + r + hiHalf * 8;
    for (int t = 0; t < 4; ++t) {
      float val = fmaxf(acc2[t][r] + fb2[t * 16 + col16], 0.0f);
      out[(size_t)node * 64 + t * 16 + col16] = val;
    }
  }
}

// ---------------------------------------------------------------------------
extern "C" void kernel_launch(void* const* d_in, const int* in_sizes, int n_in,
                              void* d_out, int out_size, void* d_ws, size_t ws_size,
                              hipStream_t stream) {
  const float* u   = (const float*)d_in[0];
  const float* v   = (const float*)d_in[1];
  const float* ev  = (const float*)d_in[2];
  const int*   idxV = (const int*)d_in[3];
  const int*   idxU = (const int*)d_in[4];
  const float* gw1 = (const float*)d_in[5];
  const float* gb1 = (const float*)d_in[6];
  const float* gw2 = (const float*)d_in[7];
  const float* gb2 = (const float*)d_in[8];
  const float* fw1 = (const float*)d_in[9];
  const float* fb1 = (const float*)d_in[10];
  const float* fw2 = (const float*)d_in[11];
  const float* fb2 = (const float*)d_in[12];

  uint32_t* ws = (uint32_t*)d_ws;     // swizzled bf16 weights (104 KB)
  float* out = (float*)d_out;         // doubles as agg buffer, then final out

  // 1) swizzle + bf16-convert all weights (26624 dword pairs)
  convert_weights<<<(26624 + 255) / 256, 256, 0, stream>>>(gw1, gw2, fw1, fw2, ws);
  // 2) zero agg (lives in d_out)
  zero_f32<<<(U_N * 64 + 255) / 256, 256, 0, stream>>>(out, U_N * 64);
  // 3) per-edge MLP + atomic scatter-sum: 800000/16 tiles, 4 waves/block
  edge_mlp<<<E_N / 64, 128, 0, stream>>>(u, v, ev, idxV, idxU,
                                         ws, ws + 10240, gb1, gb2, out);
  // 4) per-node MLP: 50000/16 = 3125 tiles, 4 waves/block
  node_mlp<<<(3125 + 3) / 4, 128, 0, stream>>>(u, out, ws + 14336, ws + 22528,
                                               fb1, fb2, out);
}
